// DSRA_Chunk_Layer_41532333752796
// MI455X (gfx1250) — matrix operand order repaired and checked by reading, not compiled, not run on hardware
//
#include <hip/hip_runtime.h>

typedef __bf16 bf16_t;
typedef __attribute__((ext_vector_type(8)))  __bf16 v8bf;
typedef __attribute__((ext_vector_type(16))) __bf16 v16bf;
typedef __attribute__((ext_vector_type(8)))  float  v8f;

typedef __attribute__((ext_vector_type(4))) unsigned int tdm_v4u;
typedef __attribute__((ext_vector_type(8))) int          tdm_v8i;
typedef __attribute__((ext_vector_type(4))) int          tdm_v4i;

static constexpr int BB = 4;
static constexpr int TT = 4096;
static constexpr int DD = 1024;
static constexpr int KK = 512;   // slots
static constexpr int KR = 16;

#define SHUF16(lo, hi) __builtin_shufflevector((lo), (hi), 0,1,2,3,4,5,6,7,8,9,10,11,12,13,14,15)

// A-fragment per-lane pattern: K {0..15} (lanes 0-15) / {16..31} (lanes 16-31)
__device__ __forceinline__ v16bf ld16_gap(const bf16_t* p) {
    v8bf lo = *(const v8bf*)p;
    v8bf hi = *(const v8bf*)(p + 8);
    return SHUF16(lo, hi);
}
// B-fragment per-lane pattern: 16 contiguous K values
__device__ __forceinline__ v16bf ld16_ct(const bf16_t* p) {
    v8bf lo = *(const v8bf*)p;
    v8bf hi = *(const v8bf*)(p + 8);
    return SHUF16(lo, hi);
}
__device__ __forceinline__ v8f wmma_bf16(v16bf a, v16bf b, v8f c) {
    return __builtin_amdgcn_wmma_f32_16x16x32_bf16(false, a, false, b, (short)0, c, false, false);
}

// --- CDNA5 async copy global->LDS (ASYNCcnt-tracked) ---
__device__ __forceinline__ uint32_t lds_off32(const void* p) {
    // LDS aperture maps addr[31:0] directly to the LDS byte address (ISA ch.10.2)
    return (uint32_t)(size_t)p;
}
__device__ __forceinline__ void async_copy_b128(uint32_t ldsoff, const bf16_t* g) {
    asm volatile("global_load_async_to_lds_b128 %0, %1, off" :: "v"(ldsoff), "v"(g) : "memory");
}
__device__ __forceinline__ void wait_async1() { asm volatile("s_wait_asynccnt 0x1" ::: "memory"); }
__device__ __forceinline__ void wait_async0() { asm volatile("s_wait_asynccnt 0x0" ::: "memory"); }

// --- CDNA5 Tensor Data Mover: DMA a strided 2-D bf16 tile global->LDS -------
// D# per CDNA5 ISA: group0 = {flags, lds_addr, global_addr, type},
// group1 = {mask/data_size, tensor dims, tile dims, dim0 stride}. Groups 2/3 zero (2-D).
// This toolchain exposes the 6-arg builtin: (v4u, v8i, v4i, v4i, v8i, i32 cpol).
#define RFL(x) __builtin_amdgcn_readfirstlane((int)(x))
__device__ __forceinline__ void tdm_load_2d(uint32_t ldsoff, const void* gptr,
                                            uint32_t rows, uint32_t rowlen_elems,
                                            uint32_t stride_elems) {
    size_t ga = (size_t)gptr;
    uint32_t dim0 = 0x40000000u, dim1 = 0x40000000u;   // huge tensor dims: no OOB clipping
    tdm_v4u g0;
    g0[0] = (unsigned)RFL(1u);                                        // count=1, user mode
    g0[1] = (unsigned)RFL(ldsoff);                                    // lds_addr
    g0[2] = (unsigned)RFL((uint32_t)(ga & 0xFFFFFFFFu));              // global_addr[31:0]
    g0[3] = (unsigned)RFL(((uint32_t)(ga >> 32) & 0x01FFFFFFu) | (2u << 30)); // addr[56:32]|type=2
    tdm_v8i g1;
    g1[0] = RFL(1u << 16);                                            // data_size=1 (2 bytes)
    g1[1] = RFL((dim0 & 0xFFFFu) << 16);                              // tensor_dim0 lo
    g1[2] = RFL((dim0 >> 16) | ((dim1 & 0xFFFFu) << 16));             // dim0 hi | dim1 lo
    g1[3] = RFL((dim1 >> 16) | (rowlen_elems << 16));                 // dim1 hi | tile_dim0
    g1[4] = RFL(rows & 0xFFFFu);                                      // tile_dim1 (tile_dim2=0)
    g1[5] = RFL(stride_elems);                                        // tensor_dim0_stride[31:0]
    g1[6] = RFL(0);
    g1[7] = RFL(0);
    tdm_v4i z4 = {0, 0, 0, 0};
    tdm_v8i z8 = {0, 0, 0, 0, 0, 0, 0, 0};
    __builtin_amdgcn_tensor_load_to_lds(g0, g1, z4, z4, z8, 0);
}

// ---------------------------------------------------------------------------
// Generic bf16 WMMA GEMM:  C[m,n] = epi( alpha * sum_k A[m,k]*Bsrc[n,k] )
// grid: (M/64, N/128, batches); 256 threads = 8 waves; wave -> 16x64 tile.
// K-tiles double-buffered in LDS: A (64x32) via per-lane async b128 copies,
// B (128x32) via one TDM DMA per tile; WMMAs run on ds_load_b128 while the
// next tile streams in (s_wait_asynccnt / s_wait_tensorcnt partial waits).
// mode 0: C = alpha*acc + (bias?bias[n]:0)
// mode 1: C = acc + bias[n] + addmat[m*ldc+n] + rowvec[m]*colvec[n]
// mode 2: C = acc + (m==n)*1e-5
// ---------------------------------------------------------------------------
__global__ __launch_bounds__(256) void gemm_bf16_kernel(
    const bf16_t* __restrict__ A, const bf16_t* __restrict__ Bsrc, float* __restrict__ C,
    int Kd, int lda, int ldb, int ldc,
    long long aStride, long long bStride, long long cStride,
    int mode, float alpha,
    const float* __restrict__ bias, const float* __restrict__ addmat,
    const float* __restrict__ rowvec, const float* __restrict__ colvec)
{
    __shared__ bf16_t sA[2][64 * 32];
    __shared__ bf16_t sB[2][128 * 32];

    int bz = blockIdx.z;
    A    += (size_t)bz * aStride;
    Bsrc += (size_t)bz * bStride;
    C    += (size_t)bz * cStride;
    int tid = threadIdx.x;
    int w = tid >> 5, lane = tid & 31;
    int mt  = blockIdx.x * 64 + (w & 3) * 16;
    int nt  = blockIdx.y * 128 + (w >> 2) * 64;
    int ntl = (w >> 2) * 64;                 // local n base within LDS B tile

    // A-tile async-copy assignment (256 threads x 16B = 4KB tile)
    int arow = tid >> 2;                     // 0..63
    int acol = (tid & 3) * 8;                // 8-elem (16B) chunk
    const bf16_t* aSrc = A + (size_t)(blockIdx.x * 64 + arow) * lda + acol;
    uint32_t aDst[2] = { lds_off32(&sA[0][arow * 32 + acol]),
                         lds_off32(&sA[1][arow * 32 + acol]) };
    // B-tile via TDM
    const bf16_t* bTile = Bsrc + (size_t)(blockIdx.y * 128) * ldb;
    uint32_t bLds[2] = { lds_off32(&sB[0][0]), lds_off32(&sB[1][0]) };

    v8f acc[4] = {};
    int aoff = (lane & 16) ? 16 : 0;
    int boff = (lane & 16) ? 16 : 0;
    int mloc = (w & 3) * 16 + (lane & 15);   // local row in LDS A tile

    // prologue: stage first K tile
    async_copy_b128(aDst[0], aSrc);
    if (w == 0) tdm_load_2d(bLds[0], bTile, 128, 32, (uint32_t)ldb);

    auto step = [&](int cur, int kcur) {
        int knext = kcur + 32;
        if (knext < Kd) {
            async_copy_b128(aDst[cur ^ 1], aSrc + knext);
            if (w == 0) tdm_load_2d(bLds[cur ^ 1], bTile + knext, 128, 32, (uint32_t)ldb);
            if (knext + 32 < Kd) __builtin_prefetch(aSrc + knext + 32, 0, 1);
            wait_async1();                                     // current A done, next in flight
            if (w == 0) __builtin_amdgcn_s_wait_tensorcnt((short)1);  // current B done
        } else {
            wait_async0();
            if (w == 0) __builtin_amdgcn_s_wait_tensorcnt((short)0);
        }
        __syncthreads();                      // tile `cur` visible to all waves

        v16bf a = ld16_gap(&sA[cur][mloc * 32 + aoff]);
        #pragma unroll
        for (int j = 0; j < 4; ++j) {
            v16bf b = ld16_ct(&sB[cur][(ntl + j * 16 + (lane & 15)) * 32 + boff]);
            acc[j] = wmma_bf16(a, b, acc[j]);
        }
        __syncthreads();                      // done reading `cur` (next iter overwrites it)
    };

    for (int kk = 0; kk < Kd; kk += 64) {     // Kd is a multiple of 64 here
        step(0, kk);
        step(1, kk + 32);
    }

    int mh = (lane >> 4) * 8, nn = lane & 15;
    #pragma unroll
    for (int j = 0; j < 4; ++j) {
        #pragma unroll
        for (int v = 0; v < 8; ++v) {
            int gm = mt + v + mh;
            int gn = nt + j * 16 + nn;
            float val = acc[j][v] * alpha;
            if (mode == 0)      { if (bias) val += bias[gn]; }
            else if (mode == 1) { val += bias[gn] + addmat[(size_t)gm * ldc + gn] + rowvec[gm] * colvec[gn]; }
            else if (mode == 2) { if (gm == gn) val += 1e-5f; }
            C[(size_t)gm * ldc + gn] = val;
        }
    }
}

// ---------------------------------------------------------------------------
// Causal SDPA (flash): scores = Q Q^T / sqrt(D), P V -> outbuf (f32 bypass).
// grid (T/16, B), 256 threads; wave w owns d-out chunk [w*128, w*128+128).
// ---------------------------------------------------------------------------
__global__ __launch_bounds__(256) void flash_kernel(
    const bf16_t* __restrict__ Qb, const bf16_t* __restrict__ Vt, float* __restrict__ outbuf)
{
    int b  = blockIdx.y;
    int m0 = blockIdx.x * 16;
    int w  = threadIdx.x >> 5, lane = threadIdx.x & 31;

    __shared__ float  sS[16 * 32];
    __shared__ bf16_t sP[16 * 32];
    __shared__ float  sM[16], sL[16], sF[16];

    if (threadIdx.x < 16) { sM[threadIdx.x] = -1e30f; sL[threadIdx.x] = 0.f; }

    v8f acc[8] = {};
    int d0   = w * 128;
    int nn   = lane & 15;
    int mh8  = (lane >> 4) * 8;
    int aoff = (lane & 16) ? 16 : 0;
    int boff = (lane & 16) ? 16 : 0;
    const bf16_t* Qbase = Qb + (size_t)b * TT * DD;
    int mrow = m0 + nn;
    int nsb  = (m0 + 15) / 32;

    // Score-pass A fragments are s-loop invariant: hoist (4 x v16bf = 32 VGPRs)
    v16bf qa[4];
    #pragma unroll
    for (int kk = 0; kk < 4; ++kk)
        qa[kk] = ld16_gap(Qbase + (size_t)mrow * DD + d0 + kk * 32 + aoff);

    for (int sb = 0; sb <= nsb; ++sb) {
        int s0 = sb * 32;
        __syncthreads();
        for (int i = threadIdx.x; i < 512; i += 256) sS[i] = 0.f;
        __syncthreads();

        if (sb < nsb) {  // deep prefetch of next s-block operands
            __builtin_prefetch(Qbase + (size_t)(s0 + 32 + nn) * DD + d0, 0, 1);
            __builtin_prefetch(Vt + ((size_t)b * DD + d0 + nn) * TT + s0 + 32, 0, 1);
        }

        // partial scores over this wave's 128-wide d chunk
        v8f c0 = {}, c1 = {};
        #pragma unroll
        for (int kk = 0; kk < 4; ++kk) {
            int ko = d0 + kk * 32;
            v16bf b0 = ld16_ct(Qbase + (size_t)(s0 + nn)      * DD + ko + boff);
            v16bf b1 = ld16_ct(Qbase + (size_t)(s0 + 16 + nn) * DD + ko + boff);
            c0 = wmma_bf16(qa[kk], b0, c0);
            c1 = wmma_bf16(qa[kk], b1, c1);
        }
        #pragma unroll
        for (int v = 0; v < 8; ++v) {
            atomicAdd(&sS[(v + mh8) * 32 + nn],      c0[v]);
            atomicAdd(&sS[(v + mh8) * 32 + nn + 16], c1[v]);
        }
        __syncthreads();

        // wave 0: online softmax bookkeeping
        if (w == 0 && lane < 16) {
            int r = lane, t = m0 + r;
            float mold = sM[r], lold = sL[r];
            float mx = mold;
            for (int c = 0; c < 32; ++c) {
                int s = s0 + c;
                float sc = (s <= t) ? sS[r * 32 + c] * 0.03125f : -1e30f;
                sS[r * 32 + c] = sc;
                if (sc > mx) mx = sc;
            }
            float f = __expf(mold - mx);
            float l = lold * f;
            for (int c = 0; c < 32; ++c) {
                float p = __expf(sS[r * 32 + c] - mx);
                l += p;
                sP[r * 32 + c] = (bf16_t)p;
            }
            sM[r] = mx; sL[r] = l; sF[r] = f;
        }
        __syncthreads();

        // rescale accumulators, then P @ V
        #pragma unroll
        for (int v = 0; v < 8; ++v) {
            float fr = sF[v + mh8];
            #pragma unroll
            for (int j = 0; j < 8; ++j) acc[j][v] *= fr;
        }
        v16bf pf = ld16_gap(&sP[nn * 32 + aoff]);
        #pragma unroll
        for (int j = 0; j < 8; ++j) {
            int dcol = d0 + j * 16 + nn;
            v16bf bfr = ld16_ct(Vt + ((size_t)b * DD + dcol) * TT + s0 + boff);
            acc[j] = wmma_bf16(pf, bfr, acc[j]);
        }
    }
    __syncthreads();
    #pragma unroll
    for (int j = 0; j < 8; ++j) {
        #pragma unroll
        for (int v = 0; v < 8; ++v) {
            int m = m0 + v + mh8;
            int d = d0 + j * 16 + nn;
            outbuf[((size_t)b * TT + m) * DD + d] = acc[j][v] / sL[v + mh8];
        }
    }
}

// ---------------------------------------------------------------------------
// Elementwise / reduction helpers
// ---------------------------------------------------------------------------
__global__ void f32_to_bf16_kernel(const float* __restrict__ src, bf16_t* __restrict__ dst, size_t n) {
    size_t i = (size_t)blockIdx.x * 256 + threadIdx.x;
    size_t st = (size_t)gridDim.x * 256;
    for (; i < n; i += st) dst[i] = (bf16_t)src[i];
}

__global__ __launch_bounds__(256) void rownorm_q_kernel(
    const float* __restrict__ Qf, const float* __restrict__ Wm, const float* __restrict__ bm,
    bf16_t* __restrict__ Qb, bf16_t* __restrict__ Qnb, float* __restrict__ instr)
{
    int row = blockIdx.x;
    const float* q = Qf + (size_t)row * DD;
    __shared__ float ss[256], sd[256];
    float s = 0.f, d = 0.f;
    for (int i = threadIdx.x; i < DD; i += 256) { float v = q[i]; s += v * v; d += v * Wm[i]; }
    ss[threadIdx.x] = s; sd[threadIdx.x] = d;
    __syncthreads();
    for (int o = 128; o; o >>= 1) {
        if (threadIdx.x < o) { ss[threadIdx.x] += ss[threadIdx.x + o]; sd[threadIdx.x] += sd[threadIdx.x + o]; }
        __syncthreads();
    }
    float inv = 1.f / fmaxf(sqrtf(ss[0]), 1e-12f);
    for (int i = threadIdx.x; i < DD; i += 256) {
        float v = q[i];
        Qb[(size_t)row * DD + i]  = (bf16_t)v;
        Qnb[(size_t)row * DD + i] = (bf16_t)(v * inv);
    }
    if (threadIdx.x == 0) instr[row] = 1.f / (1.f + __expf(-(sd[0] + bm[0])));
}

__global__ __launch_bounds__(256) void rownorm_v_kernel(
    const float* __restrict__ Vf, bf16_t* __restrict__ Vnb)
{
    int row = blockIdx.x;
    const float* r = Vf + (size_t)row * DD;
    __shared__ float ss[256];
    float s = 0.f;
    for (int i = threadIdx.x; i < DD; i += 256) { float v = r[i]; s += v * v; }
    ss[threadIdx.x] = s;
    __syncthreads();
    for (int o = 128; o; o >>= 1) { if (threadIdx.x < o) ss[threadIdx.x] += ss[threadIdx.x + o]; __syncthreads(); }
    float inv = 1.f / fmaxf(sqrtf(ss[0]), 1e-8f);
    for (int i = threadIdx.x; i < DD; i += 256) Vnb[(size_t)row * DD + i] = (bf16_t)(r[i] * inv);
}

__global__ __launch_bounds__(256) void rownorm_s_kernel(
    const float* __restrict__ S, bf16_t* __restrict__ Sb, bf16_t* __restrict__ Snb)
{
    int k = blockIdx.x;
    const float* r = S + (size_t)k * DD;
    __shared__ float ss[256];
    float s = 0.f;
    for (int i = threadIdx.x; i < DD; i += 256) { float v = r[i]; s += v * v; }
    ss[threadIdx.x] = s;
    __syncthreads();
    for (int o = 128; o; o >>= 1) { if (threadIdx.x < o) ss[threadIdx.x] += ss[threadIdx.x + o]; __syncthreads(); }
    float inv = 1.f / fmaxf(sqrtf(ss[0]), 1e-12f);
    for (int i = threadIdx.x; i < DD; i += 256) {
        float v = r[i];
        Sb[(size_t)k * DD + i]  = (bf16_t)v;
        Snb[(size_t)k * DD + i] = (bf16_t)(v * inv);
    }
}

__global__ void split_wn_kernel(const float* __restrict__ Wn, bf16_t* __restrict__ Wn1b, float* __restrict__ wn0) {
    int k = blockIdx.x;
    const float* rowp = Wn + (size_t)k * (DD + 1);
    if (threadIdx.x == 0) wn0[k] = rowp[0];
    for (int d = threadIdx.x; d < DD; d += 256) Wn1b[(size_t)k * DD + d] = (bf16_t)rowp[1 + d];
}

__global__ __launch_bounds__(256) void novelty_kernel(const float* __restrict__ sim, float* __restrict__ nov) {
    int row = blockIdx.x;
    __shared__ float sm[256];
    float m = -1e30f;
    for (int i = threadIdx.x; i < KK; i += 256) m = fmaxf(m, sim[(size_t)row * KK + i]);
    sm[threadIdx.x] = m;
    __syncthreads();
    for (int o = 128; o; o >>= 1) { if (threadIdx.x < o) sm[threadIdx.x] = fmaxf(sm[threadIdx.x], sm[threadIdx.x + o]); __syncthreads(); }
    if (threadIdx.x == 0) nov[row] = 1.f - sm[0];
}

// one wave per row: top-16 + softmax (mode 0: read routing; mode 1: write routing)
__global__ __launch_bounds__(256) void topk_kernel(
    const float* __restrict__ logits, int* __restrict__ oidx, float* __restrict__ oval,
    const float* __restrict__ nov, float* __restrict__ mass, int mode)
{
    int wv = threadIdx.x >> 5, lane = threadIdx.x & 31;
    int row = blockIdx.x * 8 + wv;
    const float* L = logits + (size_t)row * KK;
    float vals[16];
    #pragma unroll
    for (int j = 0; j < 16; ++j) vals[j] = L[j * 32 + lane];
    unsigned taken = 0;
    __shared__ float sv[8][16];
    __shared__ int   si[8][16];
    for (int it = 0; it < 16; ++it) {
        float v = -1e30f; int g = -1;
        #pragma unroll
        for (int j = 0; j < 16; ++j)
            if (!((taken >> j) & 1u) && vals[j] > v) { v = vals[j]; g = j * 32 + lane; }
        for (int o = 16; o; o >>= 1) {
            float ov = __shfl_xor(v, o);
            int   og = __shfl_xor(g, o);
            if (ov > v) { v = ov; g = og; }
        }
        if (g >= 0 && lane == (g & 31)) taken |= 1u << (g >> 5);
        if (lane == 0) { sv[wv][it] = v; si[wv][it] = g; }
    }
    __syncthreads();
    float e = 0.f; int gi = 0;
    if (lane < 16) { e = __expf(sv[wv][lane] - sv[wv][0]); gi = si[wv][lane]; }
    float tot = e;
    for (int o = 16; o; o >>= 1) tot += __shfl_xor(tot, o);
    if (lane < 16) {
        float p = e / tot;
        if (mode == 0) {
            oidx[(size_t)row * KR + lane] = gi;
            oval[(size_t)row * KR + lane] = p;
        } else {
            int b = row / TT, t = row - b * TT;
            float nv  = fminf(fmaxf(nov[row], 0.f), 1.f);
            float wgt = __expf((float)(TT - 1 - t) * -0.010050335853f); // (1-0.01)^(T-1-t)
            float wt  = p * nv * wgt;
            oidx[(size_t)row * KR + lane] = gi;
            oval[(size_t)row * KR + lane] = wt;
            atomicAdd(&mass[b * KK + gi], wt);
        }
    }
}

__global__ __launch_bounds__(256) void scatter_vagg_kernel(
    const int* __restrict__ widx, const float* __restrict__ wval,
    const float* __restrict__ Vf, float* __restrict__ Vagg)
{
    int row = blockIdx.x;
    int b = row / TT;
    int   idx[16]; float wv_[16];
    #pragma unroll
    for (int j = 0; j < 16; ++j) { idx[j] = widx[(size_t)row * KR + j]; wv_[j] = wval[(size_t)row * KR + j]; }
    for (int d = threadIdx.x; d < DD; d += 256) {
        float v = Vf[(size_t)row * DD + d];
        #pragma unroll
        for (int j = 0; j < 16; ++j)
            atomicAdd(&Vagg[((size_t)b * KK + idx[j]) * DD + d], wv_[j] * v);
    }
}

__global__ void vagg_div_kernel(float* __restrict__ Vagg, const float* __restrict__ mass, bf16_t* __restrict__ Vaggb) {
    size_t n = (size_t)BB * KK * DD;
    size_t st = (size_t)gridDim.x * 256;
    for (size_t i = (size_t)blockIdx.x * 256 + threadIdx.x; i < n; i += st) {
        size_t bk = i / DD;
        float v = Vagg[i] / fmaxf(mass[bk], 1e-6f);
        Vagg[i] = v;
        Vaggb[i] = (bf16_t)v;
    }
}

__global__ __launch_bounds__(256) void combine_kernel(
    const int* __restrict__ ridx, const float* __restrict__ rprob,
    const float* __restrict__ S, const float* __restrict__ instr, float* __restrict__ outbuf)
{
    int row = blockIdx.x;
    int idx[16]; float p[16];
    #pragma unroll
    for (int j = 0; j < 16; ++j) { idx[j] = ridx[(size_t)row * KR + j]; p[j] = rprob[(size_t)row * KR + j]; }
    float ip = instr[row];
    for (int d = threadIdx.x; d < DD; d += 256) {
        float ctx = 0.f;
        #pragma unroll
        for (int j = 0; j < 16; ++j) ctx += p[j] * S[(size_t)idx[j] * DD + d];
        size_t o = (size_t)row * DD + d;
        outbuf[o] = ctx + ip * outbuf[o];
    }
}

// tiled transpose: src f32 [R][Cc] (+batch) -> dst bf16 [Cc][R] (+batch)
__global__ __launch_bounds__(256) void transpose_f32_bf16_kernel(
    const float* __restrict__ src, bf16_t* __restrict__ dst, int R, int Cc,
    long long sStride, long long dStride)
{
    __shared__ float tile[32][33];
    int bz = blockIdx.z;
    src += (size_t)bz * sStride;
    dst += (size_t)bz * dStride;
    int c0 = blockIdx.x * 32, r0 = blockIdx.y * 32;
    for (int i = threadIdx.y; i < 32; i += 8)
        tile[i][threadIdx.x] = src[(size_t)(r0 + i) * Cc + c0 + threadIdx.x];
    __syncthreads();
    for (int i = threadIdx.y; i < 32; i += 8)
        dst[(size_t)(c0 + i) * R + r0 + threadIdx.x] = (bf16_t)tile[threadIdx.x][i];
}

// --------------------------- Gauss-Jordan inverse ---------------------------
__global__ void gj_init_kernel(const float* __restrict__ Scov, float* __restrict__ Aug) {
    int r = blockIdx.x;
    for (int c = threadIdx.x; c < 2 * KK; c += 256)
        Aug[(size_t)r * 2 * KK + c] = (c < KK) ? Scov[(size_t)r * KK + c] : ((c - KK == r) ? 1.f : 0.f);
}
__global__ void gj_elim_kernel(float* __restrict__ Aug, int p) {
    int r = blockIdx.x;
    if (r == p) return;
    __shared__ float fsh;
    if (threadIdx.x == 0) fsh = Aug[(size_t)r * 2 * KK + p] / Aug[(size_t)p * 2 * KK + p];
    __syncthreads();
    float f = fsh;
    for (int c = threadIdx.x; c < 2 * KK; c += 256)
        Aug[(size_t)r * 2 * KK + c] -= f * Aug[(size_t)p * 2 * KK + c];
}
__global__ void gj_norm_kernel(float* __restrict__ Aug, int p) {
    __shared__ float pv;
    if (threadIdx.x == 0) pv = Aug[(size_t)p * 2 * KK + p];
    __syncthreads();
    float inv = 1.f / pv;
    for (int c = threadIdx.x; c < 2 * KK; c += 256) Aug[(size_t)p * 2 * KK + c] *= inv;
}
__global__ void extract_inv_kernel(const float* __restrict__ Aug, bf16_t* __restrict__ Invb) {
    int r = blockIdx.x;
    for (int c = threadIdx.x; c < KK; c += 256)
        Invb[(size_t)r * KK + c] = (bf16_t)Aug[(size_t)r * 2 * KK + KK + c];
}

__global__ void snext_kernel(const float* __restrict__ S, const float* __restrict__ Vagg,
                             const float* __restrict__ Vproj, float* __restrict__ Sout) {
    size_t n = (size_t)BB * KK * DD;
    size_t st = (size_t)gridDim.x * 256;
    for (size_t i = (size_t)blockIdx.x * 256 + threadIdx.x; i < n; i += st) {
        size_t kd = i % ((size_t)KK * DD);
        Sout[i] = 0.99f * S[kd] + 0.1f * (Vagg[i] - Vproj[i]);
    }
}

// ---------------------------------------------------------------------------
extern "C" void kernel_launch(void* const* d_in, const int* in_sizes, int n_in,
                              void* d_out, int out_size, void* d_ws, size_t ws_size,
                              hipStream_t stream) {
    const float* x  = (const float*)d_in[0];
    const float* S  = (const float*)d_in[1];
    const float* Wq = (const float*)d_in[2];
    const float* bq = (const float*)d_in[3];
    const float* Wv = (const float*)d_in[4];
    const float* bv = (const float*)d_in[5];
    const float* Wn = (const float*)d_in[6];
    const float* bn = (const float*)d_in[7];
    const float* Wm = (const float*)d_in[8];
    const float* bm = (const float*)d_in[9];

    float* outbuf = (float*)d_out;                       // [B,T,D]
    float* snext  = outbuf + (size_t)BB * TT * DD;       // [B,K,D]

    char* base = (char*)d_ws;
    size_t off = 0;
    auto carve = [&](size_t bytes) -> char* {
        char* p = base + off;
        off += (bytes + 255) & ~(size_t)255;
        return p;
    };

    const size_t NTD = (size_t)BB * TT * DD;   // 16,777,216
    bf16_t* xb    = (bf16_t*)carve(NTD * 2);               // aliased later by RL (same size)
    float*  RL    = (float*)xb;                            // read_logits [B,T,K] f32 == 33.5MB
    bf16_t* Wqb   = (bf16_t*)carve((size_t)DD * DD * 2);
    bf16_t* Wvb   = (bf16_t*)carve((size_t)DD * DD * 2);
    bf16_t* Sb    = (bf16_t*)carve((size_t)KK * DD * 2);
    bf16_t* Snb   = (bf16_t*)carve((size_t)KK * DD * 2);
    bf16_t* Stb   = (bf16_t*)carve((size_t)DD * KK * 2);
    bf16_t* Wn1b  = (bf16_t*)carve((size_t)KK * DD * 2);
    float*  wn0   = (float*)carve((size_t)KK * 4);
    float*  Qf    = (float*)carve(NTD * 4);                // aliased later by WL (fits)
    float*  WL    = Qf;                                    // sim / write_logits [B,T,K] f32
    float*  Vf    = (float*)carve(NTD * 4);
    bf16_t* Qb    = (bf16_t*)carve(NTD * 2);
    bf16_t* Qnb   = (bf16_t*)carve(NTD * 2);
    bf16_t* Vnb   = (bf16_t*)carve(NTD * 2);               // aliased later by Vt (same size)
    bf16_t* Vt    = Vnb;                                   // [B,D,T] bf16
    float*  nov   = (float*)carve((size_t)BB * TT * 4);
    float*  instr = (float*)carve((size_t)BB * TT * 4);
    int*    ridx  = (int*)carve((size_t)BB * TT * KR * 4);
    float*  rprob = (float*)carve((size_t)BB * TT * KR * 4);
    int*    widx  = (int*)carve((size_t)BB * TT * KR * 4);
    float*  wwt   = (float*)carve((size_t)BB * TT * KR * 4);
    float*  mass  = (float*)carve((size_t)BB * KK * 4);
    float*  Vagg  = (float*)carve((size_t)BB * KK * DD * 4);
    bf16_t* Vaggb = (bf16_t*)carve((size_t)BB * KK * DD * 2);
    float*  Scov  = (float*)carve((size_t)KK * KK * 4);
    float*  Aug   = (float*)carve((size_t)KK * 2 * KK * 4);
    bf16_t* Invb  = (bf16_t*)carve((size_t)KK * KK * 2);
    float*  SVf   = (float*)carve((size_t)BB * KK * KK * 4);
    bf16_t* SVb   = (bf16_t*)carve((size_t)BB * KK * KK * 2);
    float*  coef  = (float*)carve((size_t)BB * KK * KK * 4);
    bf16_t* coefb = (bf16_t*)carve((size_t)BB * KK * KK * 2);
    float*  Vproj = (float*)carve((size_t)BB * KK * DD * 4);

    (void)hipMemsetAsync(Vagg, 0, (size_t)BB * KK * DD * 4, stream);
    (void)hipMemsetAsync(mass, 0, (size_t)BB * KK * 4, stream);

    // --- prep / conversions ---
    f32_to_bf16_kernel<<<2048, 256, 0, stream>>>(x,  xb,  NTD);
    f32_to_bf16_kernel<<<1024, 256, 0, stream>>>(Wq, Wqb, (size_t)DD * DD);
    f32_to_bf16_kernel<<<1024, 256, 0, stream>>>(Wv, Wvb, (size_t)DD * DD);
    rownorm_s_kernel<<<KK, 256, 0, stream>>>(S, Sb, Snb);
    split_wn_kernel<<<KK, 256, 0, stream>>>(Wn, Wn1b, wn0);
    transpose_f32_bf16_kernel<<<dim3(DD / 32, KK / 32, 1), dim3(32, 8), 0, stream>>>(S, Stb, KK, DD, 0, 0);

    // --- projections: Q = x Wq^T + bq ; V = x Wv^T + bv ---
    gemm_bf16_kernel<<<dim3(BB * TT / 64, DD / 128, 1), 256, 0, stream>>>(
        xb, Wqb, Qf, DD, DD, DD, DD, 0, 0, 0, 0, 1.f, bq, nullptr, nullptr, nullptr);
    gemm_bf16_kernel<<<dim3(BB * TT / 64, DD / 128, 1), 256, 0, stream>>>(
        xb, Wvb, Vf, DD, DD, DD, DD, 0, 0, 0, 0, 1.f, bv, nullptr, nullptr, nullptr);

    rownorm_q_kernel<<<BB * TT, 256, 0, stream>>>(Qf, Wm, bm, Qb, Qnb, instr);
    rownorm_v_kernel<<<BB * TT, 256, 0, stream>>>(Vf, Vnb);

    // --- read_logits = l2(Q) l2(S)^T * sqrt(D) ; sim = l2(V) l2(S)^T ---
    gemm_bf16_kernel<<<dim3(BB * TT / 64, KK / 128, 1), 256, 0, stream>>>(
        Qnb, Snb, RL, DD, DD, DD, KK, 0, 0, 0, 0, 32.f, nullptr, nullptr, nullptr, nullptr);
    gemm_bf16_kernel<<<dim3(BB * TT / 64, KK / 128, 1), 256, 0, stream>>>(
        Vnb, Snb, WL, DD, DD, DD, KK, 0, 0, 0, 0, 1.f, nullptr, nullptr, nullptr, nullptr);
    novelty_kernel<<<BB * TT, 256, 0, stream>>>(WL, nov);

    // Vt transpose (after sim consumed Vnb: alias safe)
    transpose_f32_bf16_kernel<<<dim3(DD / 32, TT / 32, BB), dim3(32, 8), 0, stream>>>(
        Vf, Vt, TT, DD, (long long)TT * DD, (long long)DD * TT);

    // --- read routing + write logits + write routing ---
    topk_kernel<<<BB * TT / 8, 256, 0, stream>>>(RL, ridx, rprob, nullptr, nullptr, 0);
    gemm_bf16_kernel<<<dim3(BB * TT / 64, KK / 128, 1), 256, 0, stream>>>(
        Qb, Wn1b, WL, DD, DD, DD, KK, 0, 0, 0, 1, 1.f, bn, RL, nov, wn0);
    topk_kernel<<<BB * TT / 8, 256, 0, stream>>>(WL, widx, wwt, nov, mass, 1);

    // --- V_agg accumulation ---
    scatter_vagg_kernel<<<BB * TT, 256, 0, stream>>>(widx, wwt, Vf, Vagg);
    vagg_div_kernel<<<2048, 256, 0, stream>>>(Vagg, mass, Vaggb);

    // --- bypass attention (flash) + combine ---
    flash_kernel<<<dim3(TT / 16, BB), 256, 0, stream>>>(Qb, Vt, outbuf);
    combine_kernel<<<BB * TT, 256, 0, stream>>>(ridx, rprob, S, instr, outbuf);

    // --- orthogonal update: S_cov, inverse, SV, coeff, V_proj, S_next ---
    gemm_bf16_kernel<<<dim3(KK / 64, KK / 128, 1), 256, 0, stream>>>(
        Sb, Sb, Scov, DD, DD, DD, KK, 0, 0, 0, 2, 1.f, nullptr, nullptr, nullptr, nullptr);
    gj_init_kernel<<<KK, 256, 0, stream>>>(Scov, Aug);
    for (int p = 0; p < KK; ++p) {
        gj_elim_kernel<<<KK, 256, 0, stream>>>(Aug, p);
        gj_norm_kernel<<<1, 256, 0, stream>>>(Aug, p);
    }
    extract_inv_kernel<<<KK, 256, 0, stream>>>(Aug, Invb);

    gemm_bf16_kernel<<<dim3(KK / 64, KK / 128, BB), 256, 0, stream>>>(
        Vaggb, Sb, SVf, DD, DD, DD, KK,
        (long long)KK * DD, 0, (long long)KK * KK, 0, 1.f, nullptr, nullptr, nullptr, nullptr);
    f32_to_bf16_kernel<<<2048, 256, 0, stream>>>(SVf, SVb, (size_t)BB * KK * KK);
    gemm_bf16_kernel<<<dim3(KK / 64, KK / 128, BB), 256, 0, stream>>>(
        SVb, Invb, coef, KK, KK, KK, KK,
        (long long)KK * KK, 0, (long long)KK * KK, 0, 1.f, nullptr, nullptr, nullptr, nullptr);
    f32_to_bf16_kernel<<<2048, 256, 0, stream>>>(coef, coefb, (size_t)BB * KK * KK);
    gemm_bf16_kernel<<<dim3(KK / 64, DD / 128, BB), 256, 0, stream>>>(
        coefb, Stb, Vproj, KK, KK, KK, DD,
        (long long)KK * KK, 0, (long long)KK * DD, 0, 1.f, nullptr, nullptr, nullptr, nullptr);
    snext_kernel<<<4096, 256, 0, stream>>>(S, Vagg, Vproj, snext);

    (void)in_sizes; (void)n_in; (void)out_size; (void)ws_size;
}
